// GraphAttentionLayer_3204045603785
// MI455X (gfx1250) — compile-verified, matrix operand order checked by reading, not run on hardware
//
#include <hip/hip_runtime.h>
#include <hip/hip_bf16.h>

typedef __bf16 bf16;
typedef __attribute__((ext_vector_type(16))) __bf16 v16bf;
typedef __attribute__((ext_vector_type(8)))  __bf16 v8bf;
typedef __attribute__((ext_vector_type(8)))  float  v8f;
typedef __attribute__((ext_vector_type(4)))  unsigned int v4u;
typedef __attribute__((ext_vector_type(8)))  int v8i;
typedef __attribute__((ext_vector_type(4)))  int v4i;

union ABu { v16bf v; struct { v8bf lo, hi; } s; };

#define WMMA_BF16(A, B, C) \
  __builtin_amdgcn_wmma_f32_16x16x32_bf16(false, (A), false, (B), (short)0, (C), false, false)

// ---------------------------------------------------------------- constants
#define NN   1024   // nodes
#define BB   8      // batch
#define EE   512    // embed
#define HH   8      // heads
#define DD   64     // head dim
#define MM   (NN*BB) // 8192 GEMM rows

// ---------------------------------------------------------------- fp32 -> bf16
__global__ void cvt_f32_bf16(const float* __restrict__ s, bf16* __restrict__ d, int n) {
  int i = blockIdx.x * blockDim.x + threadIdx.x;
  if (i < n) d[i] = (bf16)s[i];
}

// ---------------------------------------------------------------- GEMM  Y = X @ W^T + bias
// Block = 8 waves: one 16-col W tile (TDM-staged in LDS, shared), 256 rows of X.
// Each wave: 32 rows x 16 cols (2 accumulators, shared B operand).
// mode 0: f32 out[m*EE + n]                      (final projection)
// mode 1: bf16 out[((h*BB+b)*NN + i)*DD + d]     (Q / K layout)
// mode 2: bf16 out[((h*BB+b)*DD + d)*NN + i]     (V transposed layout)
__global__ __launch_bounds__(256) void gemm_bf16(
    const bf16* __restrict__ X, const bf16* __restrict__ W,
    const float* __restrict__ bias, void* __restrict__ out, int mode) {
  __shared__ bf16 Bsm[16 * EE];                  // 16 KB: W rows [tn*16, tn*16+16)

  int tid  = threadIdx.x;
  int wid  = tid >> 5, lane = tid & 31;
  int li   = lane & 15, hi = lane >> 4;
  int tn      = blockIdx.x & (EE/16 - 1);        // 0..31 col tile (block-uniform)
  int rowbase = (blockIdx.x >> 5) * 256;         // 0..31 row group of 256

  // ---- TDM: DMA the 16x512 bf16 W tile into LDS (wave 0 issues) ----
  if (wid == 0) {
    unsigned long long ga = (unsigned long long)(uintptr_t)(W + (size_t)tn * 16 * EE);
    unsigned int la = (unsigned int)(uintptr_t)Bsm;   // flat LDS addr: low 32 bits = LDS offset
    // D# group 0: count=1 | lds_addr | 57b global_addr | type=2
    v4u g0 = { 1u, la, (unsigned int)ga,
               (unsigned int)((ga >> 32) & 0x01FFFFFFu) | (2u << 30) };
    // D# group 1: data_size=2B; tensor 512x16; tile 512x16; dim0_stride=512
    v8i g1 = { (int)(1u  << 16),   // [17:16] data_size=1 (2 bytes)
               (int)(512u << 16),  // [63:48] tensor_dim0 lo16 = 512
               (int)(16u  << 16),  // [79:64] tensor_dim0 hi=0, [95:80] tensor_dim1=16
               (int)(512u << 16),  // [111:96] tensor_dim1 hi=0, [127:112] tile_dim0=512
               (int)16,            // [143:128] tile_dim1=16, [159:144] tile_dim2=0
               (int)512,           // [191:160] tensor_dim0_stride lo32 = 512
               0, 0 };             // stride hi / dim1_stride = 0
    v4i z4 = { 0, 0, 0, 0 };
    v8i z8 = { 0, 0, 0, 0, 0, 0, 0, 0 };
    __builtin_amdgcn_tensor_load_to_lds(g0, g1, z4, z4, z8, 0);
    __builtin_amdgcn_s_wait_tensorcnt((short)0);
  }
  __syncthreads();

  const bf16* aptr0 = X + (size_t)(rowbase + wid*32 + li) * EE;
  const bf16* aptr1 = aptr0 + (size_t)16 * EE;
  const bf16* bsrow = Bsm + (size_t)li * EE;     // B operand: lane = W row (= output col)

  v8f c0 = {}, c1 = {};
#pragma unroll 4
  for (int kb = 0; kb < EE; kb += 32) {
    ABu a0, a1, b;
    a0.s.lo = *(const v8bf*)(aptr0 + kb + hi*8);
    a0.s.hi = *(const v8bf*)(aptr0 + kb + 16 + hi*8);
    a1.s.lo = *(const v8bf*)(aptr1 + kb + hi*8);
    a1.s.hi = *(const v8bf*)(aptr1 + kb + 16 + hi*8);
    b.s.lo  = *(const v8bf*)(bsrow + kb + hi*8);
    b.s.hi  = *(const v8bf*)(bsrow + kb + 16 + hi*8);
    c0 = WMMA_BF16(a0.v, b.v, c0);
    c1 = WMMA_BF16(a1.v, b.v, c1);
  }

  int colg = tn*16 + li;                         // C layout: lane -> column
  float bval = bias ? bias[colg] : 0.0f;

  v8f acc[2] = { c0, c1 };
#pragma unroll
  for (int t = 0; t < 2; ++t) {
    int mbase = rowbase + wid*32 + t*16;
    if (mode == 0) {
      float* o = (float*)out;
#pragma unroll
      for (int r = 0; r < 8; ++r) {
        int m = mbase + r + hi*8;                // C layout: vgpr r -> row r (+8 hi lanes)
        o[(size_t)m * EE + colg] = acc[t][r] + bval;
      }
    } else if (mode == 1) {
      bf16* o = (bf16*)out;
      int h = colg >> 6, d = colg & (DD-1);
#pragma unroll
      for (int r = 0; r < 8; ++r) {
        int m = mbase + r + hi*8;
        int i = m >> 3, b = m & (BB-1);          // m = i*BB + b
        o[(((size_t)h*BB + b)*NN + i)*DD + d] = (bf16)(acc[t][r] + bval);
      }
    } else {
      bf16* o = (bf16*)out;
      int h = colg >> 6, d = colg & (DD-1);
#pragma unroll
      for (int r = 0; r < 8; ++r) {
        int m = mbase + r + hi*8;
        int i = m >> 3, b = m & (BB-1);
        o[(((size_t)h*BB + b)*DD + d)*NN + i] = (bf16)(acc[t][r] + bval);
      }
    }
  }
}

// ---------------------------------------------------------------- attention
// One wave = one 16-row i-tile of one (h,b). Transposed-score flash attention:
//   S^T = K_blk @ Q_tile^T  (C layout: lane holds half-column of fixed i)
//   P^T (C layout of two tiles) == B-operand layout for O^T = V^T @ P^T.
__global__ __launch_bounds__(256) void attn_kernel(
    const bf16* __restrict__ Qb, const bf16* __restrict__ Kb,
    const bf16* __restrict__ Vt, const float* __restrict__ edges,
    const float* __restrict__ rel_bias, bf16* __restrict__ Aout) {
  int wave = (blockIdx.x * blockDim.x + threadIdx.x) >> 5;
  int lane = threadIdx.x & 31;
  int li = lane & 15, hi = lane >> 4;
  int itile = wave & (NN/16 - 1);        // 0..63
  int hb    = wave >> 6;                 // 0..63
  int h = hb >> 3, b = hb & (BB-1);
  size_t base = (size_t)hb * NN * DD;

  // B operand: Q rows (lane = column i of Q^T), split over D in two k-steps
  const bf16* qrow = Qb + base + (size_t)(itile*16 + li) * DD;
  ABu bq0, bq1;
  bq0.s.lo = *(const v8bf*)(qrow + hi*8);
  bq0.s.hi = *(const v8bf*)(qrow + 16 + hi*8);
  bq1.s.lo = *(const v8bf*)(qrow + 32 + hi*8);
  bq1.s.hi = *(const v8bf*)(qrow + 48 + hi*8);

  int icol = itile*16 + li;              // this lane's query index
  float relb = rel_bias[h];
  const float scale = 0.125f;            // 1/sqrt(64)

  float m_run = -3.0e38f, l_run = 0.0f;
  v8f ot0 = {}, ot1 = {}, ot2 = {}, ot3 = {};

  for (int jb = 0; jb < NN; jb += 32) {
    // ---- scores S^T for a 32-wide j block (two 16x16 tiles) ----
    const bf16* krow0 = Kb + base + (size_t)(jb + li) * DD;
    const bf16* krow1 = krow0 + 16*DD;
    ABu ak;
    v8f s0 = {}, s1 = {};
    ak.s.lo = *(const v8bf*)(krow0 + hi*8);
    ak.s.hi = *(const v8bf*)(krow0 + 16 + hi*8);
    s0 = WMMA_BF16(ak.v, bq0.v, s0);
    ak.s.lo = *(const v8bf*)(krow0 + 32 + hi*8);
    ak.s.hi = *(const v8bf*)(krow0 + 48 + hi*8);
    s0 = WMMA_BF16(ak.v, bq1.v, s0);
    ak.s.lo = *(const v8bf*)(krow1 + hi*8);
    ak.s.hi = *(const v8bf*)(krow1 + 16 + hi*8);
    s1 = WMMA_BF16(ak.v, bq0.v, s1);
    ak.s.lo = *(const v8bf*)(krow1 + 32 + hi*8);
    ak.s.hi = *(const v8bf*)(krow1 + 48 + hi*8);
    s1 = WMMA_BF16(ak.v, bq1.v, s1);

    // edge bias: lane's row of edges, contiguous over j
    const float* ep = edges + (size_t)icol * NN + jb;
    v8f ed0 = *(const v8f*)(ep + hi*8);
    v8f ed1 = *(const v8f*)(ep + 16 + hi*8);

    float sv0[8], sv1[8], mx = -3.0e38f;
#pragma unroll
    for (int r = 0; r < 8; ++r) {
      sv0[r] = s0[r]*scale + ed0[r]*relb;
      sv1[r] = s1[r]*scale + ed1[r]*relb;
      mx = fmaxf(mx, fmaxf(sv0[r], sv1[r]));
    }
    mx = fmaxf(mx, __shfl_xor(mx, 16));          // partner lane holds other 16 j's
    float m_new = fmaxf(m_run, mx);
    float alpha = __expf(m_run - m_new);

    ABu bp;                                       // P^T straight into B-operand layout
    float rs = 0.0f;
#pragma unroll
    for (int r = 0; r < 8; ++r) {
      float p0 = __expf(sv0[r] - m_new);
      float p1 = __expf(sv1[r] - m_new);
      rs += p0 + p1;
      bp.s.lo[r] = (bf16)p0;
      bp.s.hi[r] = (bf16)p1;
    }
    rs += __shfl_xor(rs, 16);
    l_run = l_run * alpha + rs;
    m_run = m_new;
#pragma unroll
    for (int r = 0; r < 8; ++r) { ot0[r]*=alpha; ot1[r]*=alpha; ot2[r]*=alpha; ot3[r]*=alpha; }

    // ---- O^T += V^T @ P^T, 4 d-chunks of 16 ----
    const bf16* vrow = Vt + (size_t)hb * DD * NN + (size_t)li * NN + jb;
    ABu av;
    av.s.lo = *(const v8bf*)(vrow + hi*8);
    av.s.hi = *(const v8bf*)(vrow + 16 + hi*8);
    ot0 = WMMA_BF16(av.v, bp.v, ot0);
    vrow += (size_t)16 * NN;
    av.s.lo = *(const v8bf*)(vrow + hi*8);
    av.s.hi = *(const v8bf*)(vrow + 16 + hi*8);
    ot1 = WMMA_BF16(av.v, bp.v, ot1);
    vrow += (size_t)16 * NN;
    av.s.lo = *(const v8bf*)(vrow + hi*8);
    av.s.hi = *(const v8bf*)(vrow + 16 + hi*8);
    ot2 = WMMA_BF16(av.v, bp.v, ot2);
    vrow += (size_t)16 * NN;
    av.s.lo = *(const v8bf*)(vrow + hi*8);
    av.s.hi = *(const v8bf*)(vrow + 16 + hi*8);
    ot3 = WMMA_BF16(av.v, bp.v, ot3);
  }

  float inv = 1.0f / l_run;
  // store O as [m = i*BB+b][e = h*DD+d], bf16, ready as GEMM A-matrix.
  // C layout: lane = column i, vgpr r = row d (consecutive -> contiguous stores)
  bf16* obase = Aout + ((size_t)icol*BB + b)*EE + h*DD + hi*8;
  v8bf st;
#pragma unroll
  for (int r = 0; r < 8; ++r) st[r] = (bf16)(ot0[r]*inv);
  *(v8bf*)(obase +  0) = st;
#pragma unroll
  for (int r = 0; r < 8; ++r) st[r] = (bf16)(ot1[r]*inv);
  *(v8bf*)(obase + 16) = st;
#pragma unroll
  for (int r = 0; r < 8; ++r) st[r] = (bf16)(ot2[r]*inv);
  *(v8bf*)(obase + 32) = st;
#pragma unroll
  for (int r = 0; r < 8; ++r) st[r] = (bf16)(ot3[r]*inv);
  *(v8bf*)(obase + 48) = st;
}

// ---------------------------------------------------------------- launch
extern "C" void kernel_launch(void* const* d_in, const int* in_sizes, int n_in,
                              void* d_out, int out_size, void* d_ws, size_t ws_size,
                              hipStream_t stream) {
  const float* nodes = (const float*)d_in[0];
  const float* edges = (const float*)d_in[1];
  const float* Wq    = (const float*)d_in[2];
  const float* bq    = (const float*)d_in[3];
  const float* Wk    = (const float*)d_in[4];
  const float* bk    = (const float*)d_in[5];
  const float* Wv    = (const float*)d_in[6];
  const float* bv    = (const float*)d_in[7];
  const float* relb  = (const float*)d_in[8];
  const float* Wo    = (const float*)d_in[9];
  const float* bo    = (const float*)d_in[10];

  // workspace layout (bf16), ~42 MB total
  bf16* Xb   = (bf16*)d_ws;                       // [MM, EE]
  bf16* Wqb  = Xb  + (size_t)MM*EE;
  bf16* Wkb  = Wqb + (size_t)EE*EE;
  bf16* Wvb  = Wkb + (size_t)EE*EE;
  bf16* Wob  = Wvb + (size_t)EE*EE;
  bf16* Qb   = Wob + (size_t)EE*EE;               // [HH*BB, NN, DD]
  bf16* Kb   = Qb  + (size_t)HH*BB*NN*DD;
  bf16* Vt   = Kb  + (size_t)HH*BB*NN*DD;         // [HH*BB, DD, NN]
  bf16* Aout = Vt  + (size_t)HH*BB*NN*DD;         // [MM, EE]

  cvt_f32_bf16<<<(MM*EE)/256, 256, 0, stream>>>(nodes, Xb, MM*EE);
  cvt_f32_bf16<<<(EE*EE)/256, 256, 0, stream>>>(Wq, Wqb, EE*EE);
  cvt_f32_bf16<<<(EE*EE)/256, 256, 0, stream>>>(Wk, Wkb, EE*EE);
  cvt_f32_bf16<<<(EE*EE)/256, 256, 0, stream>>>(Wv, Wvb, EE*EE);
  cvt_f32_bf16<<<(EE*EE)/256, 256, 0, stream>>>(Wo, Wob, EE*EE);

  const int gemm_blocks = (EE/16) * (MM/256);     // 32 col tiles x 32 row groups = 1024
  gemm_bf16<<<gemm_blocks, 256, 0, stream>>>(Xb, Wqb, bq, Qb, 1);
  gemm_bf16<<<gemm_blocks, 256, 0, stream>>>(Xb, Wkb, bk, Kb, 1);
  gemm_bf16<<<gemm_blocks, 256, 0, stream>>>(Xb, Wvb, bv, Vt, 2);

  attn_kernel<<<(HH*BB)*(NN/16)/8, 256, 0, stream>>>(Qb, Kb, Vt, edges, relb, Aout);

  gemm_bf16<<<gemm_blocks, 256, 0, stream>>>(Aout, Wob, bo, d_out, 0);
}
